// SlicedLMHead_30313879175308
// MI455X (gfx1250) — matrix operand order repaired and checked by reading, not compiled, not run on hardware
//
#include <hip/hip_runtime.h>
#include <hip/hip_bf16.h>

typedef __attribute__((ext_vector_type(16))) __bf16 v16bf;
typedef __attribute__((ext_vector_type(8)))  __bf16 v8bf;
typedef __attribute__((ext_vector_type(8)))  float  v8f;

#define IN_C     4096
#define OUT_C    32000
#define M_TOTAL  2048
#define BM       128
#define BN       128
#define BK       32
#define LDS_STRIDE 40       // 32 + 8 pad halves: 80B row stride (20 banks) -> conflict-free, 16B aligned
#define ABUF_HALVES (BM * LDS_STRIDE)   // 5120 halves per buffer

__global__ __launch_bounds__(256)
void lmhead_wmma_bf16(const float* __restrict__ A,     // [2048, 4096] hidden
                      const float* __restrict__ W,     // [32000, 4096] weight
                      const float* __restrict__ bias,  // [32000]
                      float* __restrict__ out)         // [2048, 32000]
{
    // double-buffered LDS: [A0 | A1 | B0 | B1], each 5120 halves (20KB per matrix pair)
    __shared__ __align__(16) unsigned short smem[4 * ABUF_HALVES];

    const int tid    = threadIdx.x;
    const int lane   = tid & 31;
    const int wave   = tid >> 5;
    const int lane16 = lane & 15;
    const int laneHi = lane >> 4;           // 0 or 1

    // blockIdx.x = M tile (fastest) so the 16 sharers of a W stripe are launch-adjacent (L2 reuse)
    const int tileM = blockIdx.x * BM;
    const int tileN = blockIdx.y * BN;

    const int waveM = (wave >> 2) * 64;     // 0 or 64
    const int waveN = (wave & 3) * 32;      // 0,32,64,96

    // global->LDS staging: each thread moves 16 floats of A and 16 of W per K-step
    const int ldRow = tid >> 1;             // 0..127
    const int ldK   = (tid & 1) << 4;       // 0 or 16

    float4 ra[4], rb[4];

    v8f acc[4][2];
    #pragma unroll
    for (int i = 0; i < 4; ++i)
        #pragma unroll
        for (int j = 0; j < 2; ++j)
            #pragma unroll
            for (int e = 0; e < 8; ++e)
                acc[i][j][e] = 0.0f;

    auto global_load = [&](int kt) {
        const float4* pA = (const float4*)(A + (size_t)(tileM + ldRow) * IN_C + kt * BK + ldK);
        ra[0] = pA[0]; ra[1] = pA[1]; ra[2] = pA[2]; ra[3] = pA[3];
        const float4* pW = (const float4*)(W + (size_t)(tileN + ldRow) * IN_C + kt * BK + ldK);
        rb[0] = pW[0]; rb[1] = pW[1]; rb[2] = pW[2]; rb[3] = pW[3];
    };

    auto lds_store = [&](int buf) {
        const float* fa = (const float*)ra;
        const float* fb = (const float*)rb;
        v8bf a0, a1, b0, b1;
        #pragma unroll
        for (int i = 0; i < 8; ++i) {
            // native float->bf16 conversion (RNE) -> hardware v_cvt (packs pairs)
            a0[i] = (__bf16)fa[i];
            a1[i] = (__bf16)fa[8 + i];
            b0[i] = (__bf16)fb[i];
            b1[i] = (__bf16)fb[8 + i];
        }
        unsigned short* dA = smem + buf * ABUF_HALVES + ldRow * LDS_STRIDE + ldK;
        *(v8bf*)(dA)     = a0;
        *(v8bf*)(dA + 8) = a1;
        unsigned short* dB = smem + (2 + buf) * ABUF_HALVES + ldRow * LDS_STRIDE + ldK;
        *(v8bf*)(dB)     = b0;
        *(v8bf*)(dB + 8) = b1;
    };

    auto compute = [&](int buf) {
        const unsigned short* sA = smem + buf * ABUF_HALVES;
        const unsigned short* sB = smem + (2 + buf) * ABUF_HALVES;
        v16bf afr[4], bfr[2];

        // A 16x32 fragment: lane%16 = M; lanes 0-15 hold K 0-7 & 16-23, lanes 16-31 hold K 8-15 & 24-31
        #pragma unroll
        for (int ms = 0; ms < 4; ++ms) {
            const unsigned short* p = sA + (waveM + ms * 16 + lane16) * LDS_STRIDE + laneHi * 8;
            v8bf lo = *(const v8bf*)(p);
            v8bf hi = *(const v8bf*)(p + 16);
            afr[ms] = __builtin_shufflevector(lo, hi, 0,1,2,3,4,5,6,7,8,9,10,11,12,13,14,15);
        }
        // B 32x16 fragment: lane%16 = N; lanes 0-15 hold K 0-7 & 16-23, lanes 16-31 hold K 8-15 & 24-31
        #pragma unroll
        for (int ns = 0; ns < 2; ++ns) {
            const unsigned short* p = sB + (waveN + ns * 16 + lane16) * LDS_STRIDE + laneHi * 8;
            v8bf lo = *(const v8bf*)(p);
            v8bf hi = *(const v8bf*)(p + 16);
            bfr[ns] = __builtin_shufflevector(lo, hi, 0,1,2,3,4,5,6,7,8,9,10,11,12,13,14,15);
        }
        #pragma unroll
        for (int ms = 0; ms < 4; ++ms)
            #pragma unroll
            for (int ns = 0; ns < 2; ++ns)
                acc[ms][ns] = __builtin_amdgcn_wmma_f32_16x16x32_bf16(
                    false, afr[ms], false, bfr[ns],
                    (short)0, acc[ms][ns], false, false);
    };

    // software pipeline: prologue fills buffer 0
    global_load(0);
    lds_store(0);
    __syncthreads();

    const int KT = IN_C / BK;   // 128
    for (int kt = 0; kt < KT; ++kt) {
        const int cur = kt & 1;
        if (kt + 1 < KT) global_load(kt + 1);   // hide global latency under WMMA
        compute(cur);
        if (kt + 1 < KT) lds_store(cur ^ 1);
        __syncthreads();
    }

    // epilogue: C/D layout -> M = r + 8*laneHi, N = lane16; add bias, fp32 store
    #pragma unroll
    for (int ms = 0; ms < 4; ++ms) {
        const int mrow = tileM + waveM + ms * 16 + laneHi * 8;
        #pragma unroll
        for (int ns = 0; ns < 2; ++ns) {
            const int ncol = tileN + waveN + ns * 16 + lane16;
            const float bv = bias[ncol];
            float* o = out + (size_t)mrow * OUT_C + ncol;
            #pragma unroll
            for (int r = 0; r < 8; ++r)
                o[(size_t)r * OUT_C] = acc[ms][ns][r] + bv;
        }
    }
}

extern "C" void kernel_launch(void* const* d_in, const int* in_sizes, int n_in,
                              void* d_out, int out_size, void* d_ws, size_t ws_size,
                              hipStream_t stream) {
    (void)in_sizes; (void)n_in; (void)d_ws; (void)ws_size; (void)out_size;
    const float* hidden = (const float*)d_in[0];   // [2, 1024, 4096] fp32
    const float* weight = (const float*)d_in[1];   // [32000, 4096] fp32
    const float* bias   = (const float*)d_in[2];   // [32000] fp32
    // d_in[3] = split_num: split over K is a mathematical no-op -> fused single GEMM
    float* out = (float*)d_out;                    // [2, 1024, 32000] fp32

    dim3 grid(M_TOTAL / BM, OUT_C / BN);           // (16, 250)
    dim3 block(256);                               // 8 wave32s
    lmhead_wmma_bf16<<<grid, block, 0, stream>>>(hidden, weight, bias, out);
}
